// LSTM_71528385347790
// MI455X (gfx1250) — matrix operand order repaired and checked by reading, not compiled, not run on hardware
//
#include <hip/hip_runtime.h>
#include <stdint.h>

#define B_   256
#define T_   512
#define D_   256
#define H_   1024
#define C_   128
#define N4_  4096   // 4*H
#define KT_  1280   // D + H

typedef __attribute__((ext_vector_type(16))) __bf16 v16bf;
typedef __attribute__((ext_vector_type(8)))  float  v8f;

union FragU { struct { uint4 lo, hi; } u; v16bf v; };

__device__ __forceinline__ uint16_t f2bf(float f) {
  uint32_t u = __float_as_uint(f);
  uint32_t r = u + 0x7FFFu + ((u >> 16) & 1u);
  return (uint16_t)(r >> 16);
}

__device__ __forceinline__ float sigm_f(float x) { return 1.f / (1.f + __expf(-x)); }
__device__ __forceinline__ float tanh_f(float x) { return 1.f - 2.f / (__expf(2.f * x) + 1.f); }

// ---------------- setup kernels ----------------

__global__ void convert_f32_bf16(const float* __restrict__ src,
                                 uint16_t* __restrict__ dst, int n) {
  int i = blockIdx.x * blockDim.x + threadIdx.x;
  if (i < n) dst[i] = f2bf(src[i]);
}

// Column order of WcatT (gate-fused tiles): for n in [0,4096):
//   hblk = n>>6, gate = (n>>4)&3, h = hblk*16 + (n&15)
// so every 64-column tile holds all 4 gates of 16 consecutive h's.
// WcatT[n][k] = (k<256 ? W{gate}x[k][h] : W{gate}h[k-256][h]) in bf16, [N][K].
__global__ void pack_weights(const float* __restrict__ wgx, const float* __restrict__ wgh, const float* __restrict__ bg,
                             const float* __restrict__ wix, const float* __restrict__ wih, const float* __restrict__ bi_,
                             const float* __restrict__ wfx, const float* __restrict__ wfh, const float* __restrict__ bf_,
                             const float* __restrict__ wox, const float* __restrict__ woh, const float* __restrict__ bo_,
                             uint16_t* __restrict__ wcatT, float* __restrict__ bcat) {
  int idx = blockIdx.x * blockDim.x + threadIdx.x;
  if (idx >= N4_ * KT_) return;
  int n = idx / KT_;
  int k = idx - n * KT_;
  int gate = (n >> 4) & 3;
  int h    = ((n >> 6) << 4) | (n & 15);
  const float* wx = (gate == 0) ? wgx : (gate == 1) ? wix : (gate == 2) ? wfx : wox;
  const float* wh = (gate == 0) ? wgh : (gate == 1) ? wih : (gate == 2) ? wfh : woh;
  float v = (k < D_) ? wx[k * H_ + h] : wh[(k - D_) * H_ + h];
  wcatT[idx] = f2bf(v);
  if (k == 0) {
    const float* bb = (gate == 0) ? bg : (gate == 1) ? bi_ : (gate == 2) ? bf_ : bo_;
    bcat[n] = bb[h];
  }
}

// WphT[c][k] = Wph[k][c] (plain layout for the output projection)
__global__ void pack_wph(const float* __restrict__ wph, uint16_t* __restrict__ wphT) {
  int i = blockIdx.x * blockDim.x + threadIdx.x;
  if (i >= C_ * H_) return;
  int c = i >> 10;
  int h = i & (H_ - 1);
  wphT[i] = f2bf(wph[h * C_ + c]);
}

__global__ void init_state(float* __restrict__ cst, uint16_t* __restrict__ h0) {
  int i = blockIdx.x * blockDim.x + threadIdx.x;
  if (i >= B_ * H_) return;
  cst[i] = 0.f;
  h0[i] = 0;
}

// ---------------- fused GEMM (+optional LSTM epilogue) ----------------
// 64 threads = 2 waves. Block tile 128x64; each wave 64x64 = 4x4 accumulators,
// 16 WMMA per 32-wide K slice. K loop split in two straight-line phases
// (a0 = x_t part, a1 = h part), LDS A tile double-buffered -> 1 barrier/slice.
// FUSED: the 4 N-subtiles are g,i,f,o of lane's h column; LSTM update done
// entirely in-register, writing c (fp32, exclusive RMW) and h (bf16).

template <bool FUSED>
__global__ __launch_bounds__(64) void gemm_core(
    const uint16_t* __restrict__ a0, int lda0, int K0,
    const uint16_t* __restrict__ a1, int lda1,
    int Ktot,
    const uint16_t* __restrict__ wT, int ldw,
    const float* __restrict__ bias,
    float* __restrict__ cOrOut, uint16_t* __restrict__ hOut, int ldo) {

  __shared__ __align__(16) uint16_t ldsA[2][128 * 32];

  const int tid    = threadIdx.x;        // 0..63
  const int lane   = tid & 31;
  const int wave_m = tid >> 5;           // 0..1
  const int l15    = lane & 15;
  const int hi16   = (lane >> 4) & 1;

  const int rowTile = blockIdx.y * 128;
  const int colTile = blockIdx.x * 64;

  const v8f vzero = {0.f, 0.f, 0.f, 0.f, 0.f, 0.f, 0.f, 0.f};
  v8f acc[4][4];
#pragma unroll
  for (int mt = 0; mt < 4; ++mt)
#pragma unroll
    for (int nt = 0; nt < 4; ++nt) acc[mt][nt] = vzero;

  // B-fragment row pointers: lane holds col n, K in 8-wide chunks {8h, 16+8h} per half-wave
  const uint16_t* wRow[4];
#pragma unroll
  for (int nt = 0; nt < 4; ++nt)
    wRow[nt] = wT + (size_t)(colTile + nt * 16 + l15) * ldw + hi16 * 8;

  const int c0    = hi16 * 8;
  const int mbase = wave_m * 64;
  const int mr0 = (mbase + 0 * 16 + l15) * 32;
  const int mr1 = (mbase + 1 * 16 + l15) * 32;
  const int mr2 = (mbase + 2 * 16 + l15) * 32;
  const int mr3 = (mbase + 3 * 16 + l15) * 32;

  auto slice = [&](const uint16_t* sA0, const uint16_t* sA1, int kw, int p) {
    uint4 d0 = ((const uint4*)sA0)[0];
    uint4 d1 = ((const uint4*)sA0)[1];
    uint4 d2 = ((const uint4*)sA0)[2];
    uint4 d3 = ((const uint4*)sA0)[3];
    uint4 e0 = ((const uint4*)sA1)[0];
    uint4 e1 = ((const uint4*)sA1)[1];
    uint4 e2 = ((const uint4*)sA1)[2];
    uint4 e3 = ((const uint4*)sA1)[3];

    uint16_t* lds = &ldsA[p][0];
    *((uint4*)&lds[tid * 32 + 0])  = d0;
    *((uint4*)&lds[tid * 32 + 8])  = d1;
    *((uint4*)&lds[tid * 32 + 16]) = d2;
    *((uint4*)&lds[tid * 32 + 24]) = d3;
    *((uint4*)&lds[(tid + 64) * 32 + 0])  = e0;
    *((uint4*)&lds[(tid + 64) * 32 + 8])  = e1;
    *((uint4*)&lds[(tid + 64) * 32 + 16]) = e2;
    *((uint4*)&lds[(tid + 64) * 32 + 24]) = e3;
    __syncthreads();   // WAR on the *other* buffer is covered by next barrier

    FragU fb[4];
#pragma unroll
    for (int nt = 0; nt < 4; ++nt) {
      fb[nt].u.lo = ((const uint4*)(wRow[nt] + kw))[0];
      fb[nt].u.hi = ((const uint4*)(wRow[nt] + kw))[2];
    }

    // rotate two A-fragment registers so ds_load overlaps WMMA issue
    FragU fa0, fa1;
    fa0.u.lo = *((const uint4*)&lds[mr0 + c0]);
    fa0.u.hi = *((const uint4*)&lds[mr0 + c0 + 16]);
    fa1.u.lo = *((const uint4*)&lds[mr1 + c0]);
    fa1.u.hi = *((const uint4*)&lds[mr1 + c0 + 16]);
#pragma unroll
    for (int nt = 0; nt < 4; ++nt)
      acc[0][nt] = __builtin_amdgcn_wmma_f32_16x16x32_bf16(
          false, fa0.v, false, fb[nt].v, (short)0, acc[0][nt], false, false);
    fa0.u.lo = *((const uint4*)&lds[mr2 + c0]);
    fa0.u.hi = *((const uint4*)&lds[mr2 + c0 + 16]);
#pragma unroll
    for (int nt = 0; nt < 4; ++nt)
      acc[1][nt] = __builtin_amdgcn_wmma_f32_16x16x32_bf16(
          false, fa1.v, false, fb[nt].v, (short)0, acc[1][nt], false, false);
    fa1.u.lo = *((const uint4*)&lds[mr3 + c0]);
    fa1.u.hi = *((const uint4*)&lds[mr3 + c0 + 16]);
#pragma unroll
    for (int nt = 0; nt < 4; ++nt)
      acc[2][nt] = __builtin_amdgcn_wmma_f32_16x16x32_bf16(
          false, fa0.v, false, fb[nt].v, (short)0, acc[2][nt], false, false);
#pragma unroll
    for (int nt = 0; nt < 4; ++nt)
      acc[3][nt] = __builtin_amdgcn_wmma_f32_16x16x32_bf16(
          false, fa1.v, false, fb[nt].v, (short)0, acc[3][nt], false, false);
  };

  // phase 0: K in [0, K0) from a0  (straight-line, no per-slice branch)
  {
    const uint16_t* pa = a0 + (size_t)(rowTile + tid) * lda0;
    const uint16_t* pb = a0 + (size_t)(rowTile + tid + 64) * lda0;
    for (int k = 0; k < K0; k += 32)
      slice(pa + k, pb + k, k, (k >> 5) & 1);
  }
  // phase 1: K in [K0, Ktot) from a1
  {
    const uint16_t* pa = a1 + (size_t)(rowTile + tid) * lda1;
    const uint16_t* pb = a1 + (size_t)(rowTile + tid + 64) * lda1;
    for (int k = K0; k < Ktot; k += 32)
      slice(pa + (k - K0), pb + (k - K0), k, (k >> 5) & 1);
  }

  if (FUSED) {
    // acc[mt][0..3] = g,i,f,o pre-activations for h column hcol
    const int hcol = blockIdx.x * 16 + l15;
    const float bgv = bias[colTile + 0  + l15];
    const float biv = bias[colTile + 16 + l15];
    const float bfv = bias[colTile + 32 + l15];
    const float bov = bias[colTile + 48 + l15];
#pragma unroll
    for (int mt = 0; mt < 4; ++mt) {
      const int rbase = rowTile + wave_m * 64 + mt * 16 + hi16 * 8;
#pragma unroll
      for (int v = 0; v < 8; ++v) {
        const size_t sidx = (size_t)(rbase + v) * H_ + hcol;
        float g = tanh_f(acc[mt][0][v] + bgv);
        float i = sigm_f(acc[mt][1][v] + biv);
        float f = sigm_f(acc[mt][2][v] + bfv);
        float o = sigm_f(acc[mt][3][v] + bov);
        float cn = g * i + cOrOut[sidx] * f;
        cOrOut[sidx] = cn;
        hOut[sidx] = f2bf(tanh_f(cn) * o);
      }
    }
  } else {
#pragma unroll
    for (int nt = 0; nt < 4; ++nt) {
      const int n = colTile + nt * 16 + l15;
      const float bv = bias[n];
#pragma unroll
      for (int mt = 0; mt < 4; ++mt) {
        const int rbase = rowTile + wave_m * 64 + mt * 16 + hi16 * 8;
#pragma unroll
        for (int v = 0; v < 8; ++v)
          cOrOut[(size_t)(rbase + v) * ldo + n] = acc[mt][nt][v] + bv;
      }
    }
  }
}

// ---------------- host side ----------------

extern "C" void kernel_launch(void* const* d_in, const int* in_sizes, int n_in,
                              void* d_out, int out_size, void* d_ws, size_t ws_size,
                              hipStream_t stream) {
  const float* x   = (const float*)d_in[0];
  const float* Wgx = (const float*)d_in[1];
  const float* Wgh = (const float*)d_in[2];
  const float* bg  = (const float*)d_in[3];
  const float* Wix = (const float*)d_in[4];
  const float* Wih = (const float*)d_in[5];
  const float* bi  = (const float*)d_in[6];
  const float* Wfx = (const float*)d_in[7];
  const float* Wfh = (const float*)d_in[8];
  const float* bfv = (const float*)d_in[9];
  const float* Wox = (const float*)d_in[10];
  const float* Woh = (const float*)d_in[11];
  const float* bo  = (const float*)d_in[12];
  const float* Wph = (const float*)d_in[13];
  const float* bp  = (const float*)d_in[14];

  char* w = (char*)d_ws;
  uint16_t* xbf   = (uint16_t*)(w + 0);          // 67,108,864
  uint16_t* WcatT = (uint16_t*)(w + 67108864);   // 10,485,760
  float*    bcat  = (float*)   (w + 77594624);   //     16,384
  uint16_t* WphT  = (uint16_t*)(w + 77611008);   //    262,144
  uint16_t* hbf0  = (uint16_t*)(w + 77873152);   //    524,288
  uint16_t* hbf1  = (uint16_t*)(w + 78397440);   //    524,288
  float*    cbuf  = (float*)   (w + 78921728);   //  1,048,576
  // total: 79,970,304 bytes

  const int nx = B_ * T_ * D_;
  convert_f32_bf16<<<(nx + 255) / 256, 256, 0, stream>>>(x, xbf, nx);
  pack_weights<<<(N4_ * KT_ + 255) / 256, 256, 0, stream>>>(
      Wgx, Wgh, bg, Wix, Wih, bi, Wfx, Wfh, bfv, Wox, Woh, bo, WcatT, bcat);
  pack_wph<<<(C_ * H_ + 255) / 256, 256, 0, stream>>>(Wph, WphT);
  init_state<<<(B_ * H_ + 255) / 256, 256, 0, stream>>>(cbuf, hbf0);

  uint16_t* hbuf[2] = {hbf0, hbf1};
  dim3 gridStep(N4_ / 64, B_ / 128);   // (64, 2) = 128 workgroups
  for (int t = 0; t < T_; ++t) {
    uint16_t* hin  = hbuf[t & 1];
    uint16_t* hout = hbuf[(t + 1) & 1];
    gemm_core<true><<<gridStep, 64, 0, stream>>>(
        xbf + (size_t)t * D_, T_ * D_, D_,
        hin, H_,
        KT_, WcatT, KT_,
        bcat, cbuf, hout, 0);
  }

  // out = h_T @ Wph + bp  (h_T is in hbuf[0] since T_ is even)
  gemm_core<false><<<dim3(C_ / 64, B_ / 128), 64, 0, stream>>>(
      hbuf[0], H_, H_,
      hbuf[0], H_,
      H_, WphT, H_,
      bp, (float*)d_out, nullptr, C_);
}
